// Router_35622458753624
// MI455X (gfx1250) — matrix operand order repaired and checked by reading, not compiled, not run on hardware
//
#include <hip/hip_runtime.h>
#include <hip/hip_bf16.h>
#include <math.h>

// ---------------------------------------------------------------------------
// Router: scores = x @ w^T  (16384 x 4096) * (64 x 4096)^T -> (16384 x 64)
// + Sinkhorn aux loss (8 iters), Gumbel top-8, softmax gather.
// GEMM is HBM-bound (256MB of x @ 23.3 TB/s ~= 11us). WMMA bf16 with hi/lo
// split precision (3 WMMAs per tile) recovers ~fp32 accuracy. w is
// pre-converted to bf16 hi/lo once (L2-resident) so the hot loop does no
// redundant B-side conversion VALU work.
// ---------------------------------------------------------------------------

#define N_TOK  16384
#define DIM    4096
#define NE     64
#define TOPK   8
#define LOG_A  (-9.7040605f)   /* -ln(16384) */
#define LOG_B  (-4.1588831f)   /* -ln(64)    */

typedef __attribute__((ext_vector_type(16))) __bf16 v16bf;
typedef __attribute__((ext_vector_type(8)))  float  v8f;

__device__ __forceinline__ v8f wmma_bf16(v16bf a, v16bf b, v8f c) {
    return __builtin_amdgcn_wmma_f32_16x16x32_bf16(
        /*neg_a=*/false, a, /*neg_b=*/false, b,
        /*c_mod=*/(short)0, c, /*reuse_a=*/false, /*reuse_b=*/false);
}

// A fragment: 16x32 bf16. Lane holds K in {hl*16 .. hl*16+15}
__device__ __forceinline__ void splitA(const float* __restrict__ base, int hl,
                                       v16bf& hi, v16bf& lo) {
    const float* p0 = base + hl * 16;
    const float* p1 = base + hl * 16 + 8;
#pragma unroll
    for (int j = 0; j < 8; ++j) {
        const float x0 = p0[j];
        const __bf16 h0 = (__bf16)x0;
        hi[j]     = h0;
        lo[j]     = (__bf16)(x0 - (float)h0);
        const float x1 = p1[j];
        const __bf16 h1 = (__bf16)x1;
        hi[j + 8] = h1;
        lo[j + 8] = (__bf16)(x1 - (float)h1);
    }
}

// ---------------------------------------------------------------------------
// Kernel 0: one-time split of w (64 x 4096 f32) into bf16 hi/lo planes.
// 1 MB read, 1 MB written; both planes stay L2-resident for the GEMM.
// ---------------------------------------------------------------------------
__global__ __launch_bounds__(256) void convert_w_kernel(
    const float* __restrict__ w, __bf16* __restrict__ whi,
    __bf16* __restrict__ wlo) {
    const int i = (blockIdx.x * 256 + threadIdx.x) * 16;
#pragma unroll
    for (int j = 0; j < 16; ++j) {
        const float xv = w[i + j];
        const __bf16 h = (__bf16)xv;
        whi[i + j] = h;
        wlo[i + j] = (__bf16)(xv - (float)h);
    }
}

// ---------------------------------------------------------------------------
// Kernel 1: GEMM (WMMA, split bf16) + fused epilogue
//   - 128 blocks x 256 threads (8 waves). Block computes 128 rows x 64 experts.
//   - Wave w: rows [16w,16w+16), all 64 experts via 4 accumulators.
//   - B fragments load pre-converted bf16 directly (layout: lane=N, K=hl*16+j
//     -> 16 contiguous bf16 = one aligned 32B vector load pair).
// ---------------------------------------------------------------------------
__global__ __launch_bounds__(256) void router_gemm_topk_kernel(
    const float* __restrict__ x, const __bf16* __restrict__ whi,
    const __bf16* __restrict__ wlo, const float* __restrict__ u,
    float* __restrict__ out_c, float* __restrict__ out_idx,
    float* __restrict__ logitsT, float* __restrict__ a_dual) {
    __shared__ float s_sc[128][65];   // stride 65: conflict-free row reads

    const int wave = threadIdx.x >> 5;
    const int lane = threadIdx.x & 31;
    const int hl   = lane >> 4;
    const int l15  = lane & 15;

    const int rowg = blockIdx.x * 128 + wave * 16 + l15;   // A-row (M = l15)
    const float* xrow = x + (size_t)rowg * DIM;

    const __bf16* wh0 = whi + (size_t)(l15)      * DIM;    // B col tiles
    const __bf16* wh1 = whi + (size_t)(l15 + 16) * DIM;
    const __bf16* wh2 = whi + (size_t)(l15 + 32) * DIM;
    const __bf16* wh3 = whi + (size_t)(l15 + 48) * DIM;
    const __bf16* wl0 = wlo + (size_t)(l15)      * DIM;
    const __bf16* wl1 = wlo + (size_t)(l15 + 16) * DIM;
    const __bf16* wl2 = wlo + (size_t)(l15 + 32) * DIM;
    const __bf16* wl3 = wlo + (size_t)(l15 + 48) * DIM;

    v8f acc0 = {}, acc1 = {}, acc2 = {}, acc3 = {};

    for (int k0 = 0; k0 < DIM; k0 += 32) {
        v16bf ah, al;
        splitA(xrow + k0, hl, ah, al);
        const int bo = k0 + hl * 16;     // 32B-aligned bf16 offset

        const v16bf bh0 = *(const v16bf*)(wh0 + bo);
        const v16bf bl0 = *(const v16bf*)(wl0 + bo);
        acc0 = wmma_bf16(ah, bh0, acc0);
        acc0 = wmma_bf16(ah, bl0, acc0);
        acc0 = wmma_bf16(al, bh0, acc0);

        const v16bf bh1 = *(const v16bf*)(wh1 + bo);
        const v16bf bl1 = *(const v16bf*)(wl1 + bo);
        acc1 = wmma_bf16(ah, bh1, acc1);
        acc1 = wmma_bf16(ah, bl1, acc1);
        acc1 = wmma_bf16(al, bh1, acc1);

        const v16bf bh2 = *(const v16bf*)(wh2 + bo);
        const v16bf bl2 = *(const v16bf*)(wl2 + bo);
        acc2 = wmma_bf16(ah, bh2, acc2);
        acc2 = wmma_bf16(ah, bl2, acc2);
        acc2 = wmma_bf16(al, bh2, acc2);

        const v16bf bh3 = *(const v16bf*)(wh3 + bo);
        const v16bf bl3 = *(const v16bf*)(wl3 + bo);
        acc3 = wmma_bf16(ah, bh3, acc3);
        acc3 = wmma_bf16(ah, bl3, acc3);
        acc3 = wmma_bf16(al, bh3, acc3);
    }

    // C/D layout: c[r] = score[wave*16 + r + 8*hl][ct*16 + l15]
#pragma unroll
    for (int r = 0; r < 8; ++r) {
        const int rl = wave * 16 + r + 8 * hl;
        s_sc[rl][l15]      = acc0[r];
        s_sc[rl][16 + l15] = acc1[r];
        s_sc[rl][32 + l15] = acc2[r];
        s_sc[rl][48 + l15] = acc3[r];
    }
    __syncthreads();

    // ---- fused epilogue: one thread per row (threads 0..127) ----
    const int t = threadIdx.x;
    if (t < 128) {
        const int row = blockIdx.x * 128 + t;

        float m = -INFINITY;
#pragma unroll
        for (int e = 0; e < NE; ++e) m = fmaxf(m, s_sc[t][e]);
        float den = 0.f;
#pragma unroll
        for (int e = 0; e < NE; ++e) den += expf(s_sc[t][e] - m);
        const float lse = m + logf(den);

        // normalized logits' = s - lse_row + log_a, stored transposed [E][N]
#pragma unroll 4
        for (int e = 0; e < NE; ++e)
            logitsT[(size_t)e * N_TOK + row] = s_sc[t][e] - lse + LOG_A;
        a_dual[row] = 0.f;

        // Gumbel top-8 : g = -log(-nan_to_num(log(u)))
        float tv[TOPK]; int ti[TOPK];
#pragma unroll
        for (int k = 0; k < TOPK; ++k) { tv[k] = -INFINITY; ti[k] = 0; }
        for (int e = 0; e < NE; ++e) {
            const float uu = u[(size_t)row * NE + e];
            float lu = logf(uu);
            lu = (lu != lu) ? 0.f : lu;                      // nan_to_num
            lu = fmaxf(fminf(lu, 3.402823466e38f), -3.402823466e38f);
            const float g = -logf(-lu);
            const float v = s_sc[t][e] + g;
            if (v > tv[TOPK - 1]) {
                int j = TOPK - 1;
                while (j > 0 && v > tv[j - 1]) {
                    tv[j] = tv[j - 1]; ti[j] = ti[j - 1]; --j;
                }
                tv[j] = v; ti[j] = e;
            }
        }
#pragma unroll
        for (int k = 0; k < TOPK; ++k) {
            out_c  [(size_t)row * TOPK + k] = expf(s_sc[t][ti[k]] - lse);
            out_idx[(size_t)row * TOPK + k] = (float)ti[k];
        }
    }
}

// ---------------------------------------------------------------------------
// Sinkhorn: b_dual[e] = log_b - LSE_n(logits'[n,e] + a_dual[n])
// One block per expert column; logitsT[e*N + n] reads are contiguous.
// ---------------------------------------------------------------------------
__global__ __launch_bounds__(256) void sinkhorn_col_kernel(
    const float* __restrict__ logitsT, const float* __restrict__ a_dual,
    float* __restrict__ b_dual) {
    __shared__ float sm[256], ss[256];
    const int e = blockIdx.x;
    float m = -INFINITY, s = 0.f;
    for (int n = threadIdx.x; n < N_TOK; n += 256) {
        const float v = logitsT[(size_t)e * N_TOK + n] + a_dual[n];
        if (v > m) { s = s * expf(m - v) + 1.f; m = v; }
        else       { s += expf(v - m); }
    }
    sm[threadIdx.x] = m; ss[threadIdx.x] = s;
    __syncthreads();
    for (int off = 128; off > 0; off >>= 1) {
        if (threadIdx.x < off) {
            const float m1 = sm[threadIdx.x], s1 = ss[threadIdx.x];
            const float m2 = sm[threadIdx.x + off], s2 = ss[threadIdx.x + off];
            const float mm = fmaxf(m1, m2);
            sm[threadIdx.x] = mm;
            ss[threadIdx.x] = s1 * expf(m1 - mm) + s2 * expf(m2 - mm);
        }
        __syncthreads();
    }
    if (threadIdx.x == 0) b_dual[e] = LOG_B - (sm[0] + logf(ss[0]));
}

// a_dual[n] = log_a - LSE_e(logits'[n,e] + b_dual[e]); one thread per row.
__global__ __launch_bounds__(256) void sinkhorn_row_kernel(
    const float* __restrict__ logitsT, const float* __restrict__ b_dual,
    float* __restrict__ a_dual) {
    __shared__ float sb[NE];
    if (threadIdx.x < NE) sb[threadIdx.x] = b_dual[threadIdx.x];
    __syncthreads();
    const int n = blockIdx.x * 256 + threadIdx.x;
    float m = -INFINITY, s = 0.f;
#pragma unroll 4
    for (int e = 0; e < NE; ++e) {
        const float v = logitsT[(size_t)e * N_TOK + n] + sb[e];
        if (v > m) { s = s * expf(m - v) + 1.f; m = v; }
        else       { s += expf(v - m); }
    }
    a_dual[n] = LOG_A - (m + logf(s));
}

// aux = mean(a_dual) + mean(b_dual)
__global__ __launch_bounds__(256) void sinkhorn_reduce_kernel(
    const float* __restrict__ a_dual, const float* __restrict__ b_dual,
    float* __restrict__ out_aux) {
    __shared__ float red[256];
    float s = 0.f;
    for (int n = threadIdx.x; n < N_TOK; n += 256) s += a_dual[n];
    red[threadIdx.x] = s;
    __syncthreads();
    for (int off = 128; off > 0; off >>= 1) {
        if (threadIdx.x < off) red[threadIdx.x] += red[threadIdx.x + off];
        __syncthreads();
    }
    if (threadIdx.x == 0) {
        float sb = 0.f;
        for (int e = 0; e < NE; ++e) sb += b_dual[e];
        out_aux[0] = red[0] / (float)N_TOK + sb / (float)NE;
    }
}

// ---------------------------------------------------------------------------
extern "C" void kernel_launch(void* const* d_in, const int* in_sizes, int n_in,
                              void* d_out, int out_size, void* d_ws, size_t ws_size,
                              hipStream_t stream) {
    const float* x = (const float*)d_in[0];
    const float* w = (const float*)d_in[1];
    const float* u = (const float*)d_in[2];

    float* out_c   = (float*)d_out;                       // [16384*8]
    float* out_idx = out_c + (size_t)N_TOK * TOPK;        // [16384*8] (as float)
    float* out_aux = out_idx + (size_t)N_TOK * TOPK;      // [1]

    float*  logitsT = (float*)d_ws;                       // [64][16384] = 4 MB
    float*  a_dual  = logitsT + (size_t)NE * N_TOK;       // [16384]
    float*  b_dual  = a_dual + N_TOK;                     // [64]
    __bf16* whi     = (__bf16*)(b_dual + NE);             // [64][4096] bf16
    __bf16* wlo     = whi + (size_t)NE * DIM;             // [64][4096] bf16

    convert_w_kernel<<<NE, 256, 0, stream>>>(w, whi, wlo);

    router_gemm_topk_kernel<<<N_TOK / 128, 256, 0, stream>>>(
        x, whi, wlo, u, out_c, out_idx, logitsT, a_dual);

    for (int it = 0; it < 8; ++it) {
        sinkhorn_col_kernel<<<NE, 256, 0, stream>>>(logitsT, a_dual, b_dual);
        sinkhorn_row_kernel<<<N_TOK / 256, 256, 0, stream>>>(logitsT, b_dual, a_dual);
    }
    sinkhorn_reduce_kernel<<<1, 256, 0, stream>>>(a_dual, b_dual, out_aux);
}
